// QwenImageCrossAttention_30528627540150
// MI455X (gfx1250) — matrix operand order repaired and checked by reading, not compiled, not run on hardware
//
#include <hip/hip_runtime.h>
#include <hip/hip_bf16.h>

#define D_MODEL 3072
#define S_ALL   2560
#define S_TXTC  512
#define N_HEADS 24
#define HD      128

typedef __attribute__((ext_vector_type(16))) _Float16 v16h;
typedef __attribute__((ext_vector_type(8)))  _Float16 v8h;
typedef __attribute__((ext_vector_type(8)))  float    v8f;
typedef int v4i_vec __attribute__((vector_size(16)));   // matches builtin param type

__device__ __forceinline__ v16h cat8(v8h lo, v8h hi) {
  return __builtin_shufflevector(lo, hi, 0,1,2,3,4,5,6,7,8,9,10,11,12,13,14,15);
}
__device__ __forceinline__ v8f wmma16(v16h a, v16h b, v8f c) {
  return __builtin_amdgcn_wmma_f32_16x16x32_f16(false, a, false, b, (short)0, c, false, false);
}

// ---- CDNA5 async global->LDS path (signature from compiler diagnostic) -----
#if __has_builtin(__builtin_amdgcn_global_load_async_to_lds_b128)
#define ASYNC_LDS 1
#define ASYNC_CP16(g, l, off)                                             \
  __builtin_amdgcn_global_load_async_to_lds_b128(                         \
      (__attribute__((address_space(1))) v4i_vec*)(g),                    \
      (__attribute__((address_space(3))) v4i_vec*)(l), (off), 0)
#else
#define ASYNC_LDS 0
#endif

__device__ __forceinline__ void async_wait0() {
#if __has_builtin(__builtin_amdgcn_s_wait_asynccnt)
  __builtin_amdgcn_s_wait_asynccnt(0);
#elif ASYNC_LDS
  asm volatile("s_wait_asynccnt 0x0" ::: "memory");
#endif
}

// ---------------------------------------------------------------------------
// fp32 -> f16 elementwise convert
// ---------------------------------------------------------------------------
__global__ void cvt_f32_f16(const float* __restrict__ src, _Float16* __restrict__ dst, int n) {
  int i = blockIdx.x * blockDim.x + threadIdx.x;
  if (i < n) dst[i] = (_Float16)src[i];
}

// ---------------------------------------------------------------------------
// GEMM: C[M,3072] = A[M,3072](f16) * W[3072,3072](f32, row-major (o,d))^T + bias
// 256 thr = 8 waves (4x2), block tile 128(M) x 128(N), per-wave 32x64,
// K-step 32, double-buffered LDS; B kept fp32 in LDS, cvt at fragment build.
// ---------------------------------------------------------------------------
__global__ void __launch_bounds__(256)
gemm_wmma(const _Float16* __restrict__ A, const float* __restrict__ W,
          const float* __restrict__ bias, float* __restrict__ C) {
  __shared__ __align__(16) _Float16 aS[2][128][40];
  __shared__ __align__(16) float    bS[2][128][36];
  const int t    = threadIdx.x;
  const int lane = t & 31;
  const int wave = t >> 5;
  const int wy   = wave >> 1;          // 0..3 (M)
  const int wx   = wave & 1;           // 0..1 (N)
  const int lg   = lane >> 4;          // half-wave
  const int lr   = lane & 15;
  const int m_blk = blockIdx.x * 128;
  const int n_blk = blockIdx.y * 128;

  // staging assignment: 2 threads per row, 16 elems each
  const int srow = t >> 1;
  const int scol = (t & 1) * 16;

  auto stage = [&](int k0, int buf) {
    const _Float16* ga = &A[(size_t)(m_blk + srow) * D_MODEL + k0 + scol];
    const float*    gw = &W[(size_t)(n_blk + srow) * D_MODEL + k0 + scol];
    _Float16* la = &aS[buf][srow][scol];
    float*    lb = &bS[buf][srow][scol];
#if ASYNC_LDS
    ASYNC_CP16(ga, la, 0);
    ASYNC_CP16(ga, la, 16);
    ASYNC_CP16(gw, lb, 0);
    ASYNC_CP16(gw, lb, 16);
    ASYNC_CP16(gw, lb, 32);
    ASYNC_CP16(gw, lb, 48);
#else
    const v8h*    pa = (const v8h*)ga;
    const float4* pw = (const float4*)gw;
    v8h a0 = pa[0], a1 = pa[1];
    float4 w0 = pw[0], w1 = pw[1], w2 = pw[2], w3 = pw[3];
    *(v8h*)(la)     = a0;
    *(v8h*)(la + 8) = a1;
    float4* lb4 = (float4*)lb;
    lb4[0] = w0; lb4[1] = w1; lb4[2] = w2; lb4[3] = w3;
#endif
  };

  v8f acc[2][4];
#pragma unroll
  for (int i = 0; i < 2; i++)
#pragma unroll
    for (int j = 0; j < 4; j++) acc[i][j] = (v8f){0,0,0,0,0,0,0,0};

  stage(0, 0);
  async_wait0();
  __syncthreads();

  for (int k0 = 0; k0 < D_MODEL; k0 += 32) {
    const int cur = (k0 >> 5) & 1;
    if (k0 + 32 < D_MODEL) stage(k0 + 32, cur ^ 1);

    v16h af[2], bf[4];
#pragma unroll
    for (int i = 0; i < 2; i++) {
      int m = wy * 32 + i * 16 + lr;
      v8h lo = *(const v8h*)&aS[cur][m][lg * 16];
      v8h hi = *(const v8h*)&aS[cur][m][lg * 16 + 8];
      af[i] = cat8(lo, hi);
    }
#pragma unroll
    for (int j = 0; j < 4; j++) {
      int n = wx * 64 + j * 16 + lr;
      const float4* p = (const float4*)&bS[cur][n][lg * 16];
      float4 x0 = p[0], x1 = p[1], x2 = p[2], x3 = p[3];
      v8h h0, h1;
      h0[0]=(_Float16)x0.x; h0[1]=(_Float16)x0.y; h0[2]=(_Float16)x0.z; h0[3]=(_Float16)x0.w;
      h0[4]=(_Float16)x1.x; h0[5]=(_Float16)x1.y; h0[6]=(_Float16)x1.z; h0[7]=(_Float16)x1.w;
      h1[0]=(_Float16)x2.x; h1[1]=(_Float16)x2.y; h1[2]=(_Float16)x2.z; h1[3]=(_Float16)x2.w;
      h1[4]=(_Float16)x3.x; h1[5]=(_Float16)x3.y; h1[6]=(_Float16)x3.z; h1[7]=(_Float16)x3.w;
      bf[j] = cat8(h0, h1);
    }
#pragma unroll
    for (int i = 0; i < 2; i++)
#pragma unroll
      for (int j = 0; j < 4; j++) acc[i][j] = wmma16(af[i], bf[j], acc[i][j]);

    async_wait0();
    __syncthreads();
  }

#pragma unroll
  for (int i = 0; i < 2; i++)
#pragma unroll
    for (int j = 0; j < 4; j++)
#pragma unroll
      for (int r = 0; r < 8; r++) {
        int row = m_blk + wy * 32 + i * 16 + r + 8 * lg;
        int col = n_blk + wx * 64 + j * 16 + lr;
        C[(size_t)row * D_MODEL + col] = acc[i][j][r] + bias[col];
      }
}

// ---------------------------------------------------------------------------
// RMSNorm + interleaved RoPE on Q/K, V transpose; fp32 in, f16 out.
// grid (S_ALL, N_HEADS), block 128. Rows 0..511 are txt, rest img.
// ---------------------------------------------------------------------------
__global__ void __launch_bounds__(128)
norm_rope(const float* __restrict__ qf, const float* __restrict__ kf,
          const float* __restrict__ vf,
          const float* __restrict__ g_q,  const float* __restrict__ g_k,
          const float* __restrict__ g_aq, const float* __restrict__ g_ak,
          const float* __restrict__ img_cache, const float* __restrict__ txt_cache,
          _Float16* __restrict__ Qh, _Float16* __restrict__ Kh,
          _Float16* __restrict__ Vt) {
  const int s = blockIdx.x, h = blockIdx.y, t = threadIdx.x;
  const size_t idx = (size_t)s * D_MODEL + h * HD + t;
  float qv = qf[idx], kv = kf[idx], vv = vf[idx];

  __shared__ float red[2][128];
  red[0][t] = qv * qv; red[1][t] = kv * kv;
  __syncthreads();
  for (int off = 64; off > 0; off >>= 1) {
    if (t < off) { red[0][t] += red[0][t + off]; red[1][t] += red[1][t + off]; }
    __syncthreads();
  }
  const float rq = rsqrtf(red[0][0] * (1.0f / HD) + 1e-6f);
  const float rk = rsqrtf(red[1][0] * (1.0f / HD) + 1e-6f);

  const bool txt = (s < S_TXTC);
  const float* cache = txt ? (txt_cache + (size_t)s * HD)
                           : (img_cache + (size_t)(s - S_TXTC) * HD);
  qv *= rq * (txt ? g_aq[t] : g_q[t]);
  kv *= rk * (txt ? g_ak[t] : g_k[t]);

  const int i = t >> 1;
  const float c = cache[i], sn = cache[64 + i];
  const float qp = __shfl_xor(qv, 1);
  const float kp = __shfl_xor(kv, 1);
  const float qo = (t & 1) ? (qv * c + qp * sn) : (qv * c - qp * sn);
  const float ko = (t & 1) ? (kv * c + kp * sn) : (kv * c - kp * sn);

  Qh[idx] = (_Float16)(qo * 0.08838834764831845f);  // fold 1/sqrt(128)
  Kh[idx] = (_Float16)ko;
  Vt[(size_t)(h * HD + t) * S_ALL + s] = (_Float16)vv;   // (h*d, s)
}

// ---------------------------------------------------------------------------
// Flash attention: grid (S_ALL/128, N_HEADS), 256 thr = 8 waves x 16 q-rows.
// Q (s,h,d) f16 pre-scaled, K (s,h,d) f16, Vt (h*d, s) f16 -> O (s, h*d) f16
// ---------------------------------------------------------------------------
__global__ void __launch_bounds__(256)
attn_wmma(const _Float16* __restrict__ Q, const _Float16* __restrict__ K,
          const _Float16* __restrict__ Vt, _Float16* __restrict__ O) {
  __shared__ __align__(16) _Float16 pS[8][16][40];
  const int h    = blockIdx.y;
  const int wave = threadIdx.x >> 5;
  const int lane = threadIdx.x & 31;
  const int lg   = lane >> 4, lr = lane & 15;
  const int q0   = blockIdx.x * 128 + wave * 16;

  v16h qfr[4];
#pragma unroll
  for (int j = 0; j < 4; j++) {
    const _Float16* base = Q + (size_t)(q0 + lr) * D_MODEL + h * HD + j * 32;
    v8h lo = *(const v8h*)(base + lg * 16);
    v8h hi = *(const v8h*)(base + lg * 16 + 8);
    qfr[j] = cat8(lo, hi);
  }

  v8f acc[8];
#pragma unroll
  for (int j = 0; j < 8; j++) acc[j] = (v8f){0,0,0,0,0,0,0,0};
  float mrow[8], lrow[8];
#pragma unroll
  for (int r = 0; r < 8; r++) { mrow[r] = -1e30f; lrow[r] = 0.f; }

  for (int kc = 0; kc < S_ALL; kc += 32) {
    v8f s0 = (v8f){0,0,0,0,0,0,0,0};
    v8f s1 = (v8f){0,0,0,0,0,0,0,0};
#pragma unroll
    for (int j = 0; j < 4; j++) {
      const _Float16* kb0 = K + (size_t)(kc + lr)      * D_MODEL + h * HD + j * 32 + lg * 16;
      const _Float16* kb1 = K + (size_t)(kc + 16 + lr) * D_MODEL + h * HD + j * 32 + lg * 16;
      s0 = wmma16(qfr[j], cat8(*(const v8h*)kb0, *(const v8h*)(kb0 + 8)), s0);
      s1 = wmma16(qfr[j], cat8(*(const v8h*)kb1, *(const v8h*)(kb1 + 8)), s1);
    }
    // online softmax: per-row stats live in 16-lane halves
    float pm[8];
#pragma unroll
    for (int r = 0; r < 8; r++) pm[r] = fmaxf(s0[r], s1[r]);
#pragma unroll
    for (int m = 1; m < 16; m <<= 1)
#pragma unroll
      for (int r = 0; r < 8; r++) pm[r] = fmaxf(pm[r], __shfl_xor(pm[r], m));
    float alpha[8];
#pragma unroll
    for (int r = 0; r < 8; r++) {
      float mn = fmaxf(mrow[r], pm[r]);
      alpha[r] = __expf(mrow[r] - mn);
      mrow[r] = mn;
    }
    float ps[8];
#pragma unroll
    for (int r = 0; r < 8; r++) {
      s0[r] = __expf(s0[r] - mrow[r]);
      s1[r] = __expf(s1[r] - mrow[r]);
      ps[r] = s0[r] + s1[r];
    }
#pragma unroll
    for (int m = 1; m < 16; m <<= 1)
#pragma unroll
      for (int r = 0; r < 8; r++) ps[r] += __shfl_xor(ps[r], m);
#pragma unroll
    for (int r = 0; r < 8; r++) lrow[r] = lrow[r] * alpha[r] + ps[r];
#pragma unroll
    for (int j = 0; j < 8; j++)
#pragma unroll
      for (int r = 0; r < 8; r++) acc[j][r] *= alpha[r];

    // C-layout -> A-layout transpose of P through per-wave LDS slice
#pragma unroll
    for (int r = 0; r < 8; r++) {
      pS[wave][r + 8 * lg][lr]      = (_Float16)s0[r];
      pS[wave][r + 8 * lg][16 + lr] = (_Float16)s1[r];
    }
    v16h pf;
    {
      const _Float16* base = &pS[wave][lr][0];
      v8h lo = *(const v8h*)(base + lg * 16);
      v8h hi = *(const v8h*)(base + lg * 16 + 8);
      pf = cat8(lo, hi);
    }
#pragma unroll
    for (int j = 0; j < 8; j++) {
      const _Float16* vb = Vt + (size_t)(h * HD + j * 16 + lr) * S_ALL + kc + lg * 16;
      acc[j] = wmma16(pf, cat8(*(const v8h*)vb, *(const v8h*)(vb + 8)), acc[j]);
    }
  }

  float inv[8];
#pragma unroll
  for (int r = 0; r < 8; r++) inv[r] = 1.0f / lrow[r];
#pragma unroll
  for (int j = 0; j < 8; j++)
#pragma unroll
    for (int r = 0; r < 8; r++) {
      int row = q0 + r + 8 * lg;
      O[(size_t)row * D_MODEL + h * HD + j * 16 + lr] = (_Float16)(acc[j][r] * inv[r]);
    }
}

// ---------------------------------------------------------------------------
extern "C" void kernel_launch(void* const* d_in, const int* in_sizes, int n_in,
                              void* d_out, int out_size, void* d_ws, size_t ws_size,
                              hipStream_t stream) {
  const float* hs   = (const float*)d_in[0];
  const float* ehs  = (const float*)d_in[1];
  const float* Wq   = (const float*)d_in[2];  const float* bq   = (const float*)d_in[3];
  const float* Wk   = (const float*)d_in[4];  const float* bk   = (const float*)d_in[5];
  const float* Wv   = (const float*)d_in[6];  const float* bv   = (const float*)d_in[7];
  const float* Waq  = (const float*)d_in[8];  const float* baq  = (const float*)d_in[9];
  const float* Wak  = (const float*)d_in[10]; const float* bak  = (const float*)d_in[11];
  const float* Wav  = (const float*)d_in[12]; const float* bav  = (const float*)d_in[13];
  const float* g_q  = (const float*)d_in[14]; const float* g_k  = (const float*)d_in[15];
  const float* g_aq = (const float*)d_in[16]; const float* g_ak = (const float*)d_in[17];
  const float* Wout = (const float*)d_in[18]; const float* bout = (const float*)d_in[19];
  const float* Wadd = (const float*)d_in[20]; const float* badd = (const float*)d_in[21];
  const float* img_cache = (const float*)d_in[22];
  const float* txt_cache = (const float*)d_in[23];
  float* out = (float*)d_out;

  char* ws = (char*)d_ws;
  size_t off = 0;
  auto take = [&](size_t bytes) -> char* {
    char* p = ws + off;
    off += (bytes + 255) & ~(size_t)255;
    return p;
  };
  const size_t SD = (size_t)S_ALL * D_MODEL;
  _Float16* x_h  = (_Float16*)take(SD * 2);  // txt rows 0..511, img rows 512..
  float*    q_f  = (float*)take(SD * 4);
  float*    k_f  = (float*)take(SD * 4);
  float*    v_f  = (float*)take(SD * 4);
  _Float16* q_h  = (_Float16*)take(SD * 2);
  _Float16* k_h  = (_Float16*)take(SD * 2);
  _Float16* v_t  = (_Float16*)take(SD * 2);  // (h*d, s)
  _Float16* a_h  = (_Float16*)take(SD * 2);
  (void)ws_size; (void)in_sizes; (void)n_in; (void)out_size;

  // 1) activations fp32 -> f16 (concat order: txt first, img at row 512)
  {
    int n_img = 2048 * D_MODEL, n_txt = S_TXTC * D_MODEL;
    cvt_f32_f16<<<(n_img + 255) / 256, 256, 0, stream>>>(hs,  x_h + (size_t)S_TXTC * D_MODEL, n_img);
    cvt_f32_f16<<<(n_txt + 255) / 256, 256, 0, stream>>>(ehs, x_h, n_txt);
  }
  const _Float16* x_txt = x_h;
  const _Float16* x_img = x_h + (size_t)S_TXTC * D_MODEL;

  // 2) QKV projections (bias fused)
  dim3 blk(256);
  dim3 g_img(2048 / 128, D_MODEL / 128), g_txt(S_TXTC / 128, D_MODEL / 128);
  gemm_wmma<<<g_img, blk, 0, stream>>>(x_img, Wq,  bq,  q_f + (size_t)S_TXTC * D_MODEL);
  gemm_wmma<<<g_txt, blk, 0, stream>>>(x_txt, Waq, baq, q_f);
  gemm_wmma<<<g_img, blk, 0, stream>>>(x_img, Wk,  bk,  k_f + (size_t)S_TXTC * D_MODEL);
  gemm_wmma<<<g_txt, blk, 0, stream>>>(x_txt, Wak, bak, k_f);
  gemm_wmma<<<g_img, blk, 0, stream>>>(x_img, Wv,  bv,  v_f + (size_t)S_TXTC * D_MODEL);
  gemm_wmma<<<g_txt, blk, 0, stream>>>(x_txt, Wav, bav, v_f);

  // 3) RMSNorm + RoPE + V transpose
  norm_rope<<<dim3(S_ALL, N_HEADS), dim3(128), 0, stream>>>(
      q_f, k_f, v_f, g_q, g_k, g_aq, g_ak, img_cache, txt_cache, q_h, k_h, v_t);

  // 4) flash attention
  attn_wmma<<<dim3(S_ALL / 128, N_HEADS), dim3(256), 0, stream>>>(q_h, k_h, v_t, a_h);

  // 5) output projections: img_out first, then txt_out
  gemm_wmma<<<g_img, blk, 0, stream>>>(a_h + (size_t)S_TXTC * D_MODEL, Wout, bout, out);
  gemm_wmma<<<g_txt, blk, 0, stream>>>(a_h, Wadd, badd, out + (size_t)2048 * D_MODEL);
}
